// HighOrderTreeSequentialGCNModel_40046275068619
// MI455X (gfx1250) — compile-verified, matrix operand order checked by reading, not run on hardware
//
#include <hip/hip_runtime.h>
#include <hip/hip_bf16.h>

// ---------------- constants (match reference) ----------------
constexpr int kS  = 8;
constexpr int kN  = 32768;
constexpr int kE  = 131072;
constexpr int kB  = 32;
constexpr float kEPS = 1e-5f;
constexpr int kK1 = 160;   // 129 padded to multiple of 32

// ---------------- bf16 / WMMA helpers ----------------
typedef __attribute__((ext_vector_type(16))) __bf16 bf16x16;
typedef __attribute__((ext_vector_type(8)))  float  f32x8;

union Frag { bf16x16 v; unsigned short u[16]; };

static __device__ inline unsigned short f32_to_bf16(float f) {
  unsigned int x = __float_as_uint(f);
  x += 0x7fffu + ((x >> 16) & 1u);          // round-to-nearest-even
  return (unsigned short)(x >> 16);
}
static __device__ inline float bf16_to_f32(unsigned short u) {
  return __uint_as_float(((unsigned int)u) << 16);
}
static __device__ inline float sigmf(float x) { return 1.f / (1.f + __expf(-x)); }

// A-fragment: 16x32 (MxK) bf16. Lane L<16: row M=L, K 0-7 & 16-23;
// lane L>=16: row M=L-16, K 8-15 & 24-31. Two 16B loads per lane.
static __device__ inline bf16x16 load_afrag(const unsigned short* p, int ld) {
  const int lane = threadIdx.x & 31;
  const unsigned short* row = p + (lane & 15) * ld + ((lane >> 4) << 3);
  Frag f;
  ((uint4*)f.u)[0] = *(const uint4*)row;
  ((uint4*)f.u)[1] = *(const uint4*)(row + 16);
  return f.v;
}
// B-fragment from packed (fragment-ready) storage: 512 contiguous ushorts per
// 32(K)x16(N) tile, lane-major. Two 16B loads per lane.
static __device__ inline bf16x16 load_bfrag_packed(const unsigned short* p) {
  const uint4* q = (const uint4*)(p + ((threadIdx.x & 31) << 4));
  Frag f;
  ((uint4*)f.u)[0] = q[0];
  ((uint4*)f.u)[1] = q[1];
  return f.v;
}
static __device__ inline f32x8 wmma_bf16(bf16x16 a, bf16x16 b, f32x8 c) {
  return __builtin_amdgcn_wmma_f32_16x16x32_bf16(false, a, false, b, (short)0, c,
                                                 false, false);
}
// packed offset for element (k, n) of a K x Ncols B matrix
static __device__ inline size_t packed_off(int k, int n, int Ncols) {
  int kk = k & 31;
  int lane = (n & 15) | (((kk >> 3) & 1) << 4);
  int idx  = (kk & 7) | (((kk >> 4) & 1) << 3);
  return (size_t)(k >> 5) * ((size_t)(Ncols >> 4) * 512) +
         (size_t)(n >> 4) * 512 + (size_t)lane * 16 + idx;
}

// ---------------- weight conversion (to packed bf16 fragments) ----------------
__global__ void k_convert_pad_packed(const float* __restrict__ W,
                                     unsigned short* __restrict__ out,
                                     int Kreal, int Kpad) {
  int i = blockIdx.x * blockDim.x + threadIdx.x;
  if (i >= Kpad * 256) return;
  int k = i >> 8, n = i & 255;
  float v = (k < Kreal) ? W[k * 256 + n] : 0.f;
  out[packed_off(k, n, 256)] = f32_to_bf16(v);
}
// W:[J][K] -> packed transpose [K x J] (J multiple of 16, K multiple of 32)
__global__ void k_transpose_packed(const float* __restrict__ W,
                                   unsigned short* __restrict__ out, int J, int K) {
  int i = blockIdx.x * blockDim.x + threadIdx.x;
  if (i >= J * K) return;
  int k = i / J, j = i - k * J;
  out[packed_off(k, j, J)] = f32_to_bf16(W[j * K + k]);
}

// ---------------- feature assembly ----------------
__global__ void k_assemble(const float* __restrict__ poss, const int* __restrict__ xidx,
                           const float* __restrict__ e0, const float* __restrict__ e1,
                           const float* __restrict__ e2, const float* __restrict__ e5,
                           unsigned short* __restrict__ xb) {
  int i = blockIdx.x * blockDim.x + threadIdx.x;
  if (i >= kN * kK1) return;
  int r = i / kK1, c = i - r * kK1;
  float v = 0.f;
  if (c == 0) v = poss[r];
  else if (c < 129) {
    int s = (c - 1) >> 5, o = (c - 1) & 31;
    const float* t = (s == 0) ? e0 : (s == 1) ? e1 : (s == 2) ? e2 : e5;
    v = t[xidx[r * 4 + s] * 32 + o];
  }
  xb[(size_t)r * kK1 + c] = f32_to_bf16(v);
}

// ---------------- WMMA GEMM + fused GCN self-loop/bias epilogue ----------------
// h[M x 256] = A[M x K](bf16) @ Bpacked[K x 256](bf16)
// agg[M x 256] = h * dinv[row]^2 + bias[col]   (seed for edge scatter-add)
// Software-pipelined: next chunk's fragments are loaded before current wmma ops,
// so waits become partial (s_wait_loadcnt N) and L2 latency overlaps WMMA.
__global__ __launch_bounds__(256) void k_gemm_bf16(const unsigned short* __restrict__ A,
                                                   const unsigned short* __restrict__ Bp,
                                                   float* __restrict__ C,
                                                   float* __restrict__ agg,
                                                   const float* __restrict__ dinv,
                                                   const float* __restrict__ bias,
                                                   int K) {
  const int tid = threadIdx.x;
  const int w = tid >> 5;
  const int rowBase = blockIdx.x * 128 + (w << 4);
  const int colBase = blockIdx.y * 64;
  const unsigned short* aptr = A + (size_t)rowBase * K;
  const size_t bcol = (size_t)(colBase >> 4) * 512;

  f32x8 acc[4] = {};
  bf16x16 a = load_afrag(aptr, K);
  const unsigned short* bb = Bp + bcol;
  bf16x16 b0 = load_bfrag_packed(bb);
  bf16x16 b1 = load_bfrag_packed(bb + 512);
  bf16x16 b2 = load_bfrag_packed(bb + 1024);
  bf16x16 b3 = load_bfrag_packed(bb + 1536);

  for (int k0 = 0; k0 < K; k0 += 32) {
    bf16x16 an = a, n0 = b0, n1 = b1, n2 = b2, n3 = b3;
    if (k0 + 32 < K) {
      __builtin_prefetch(aptr + k0 + 64, 0, 1);
      an = load_afrag(aptr + k0 + 32, K);
      const unsigned short* bbn =
          Bp + (size_t)((k0 + 32) >> 5) * (16 * 512) + bcol;
      n0 = load_bfrag_packed(bbn);
      n1 = load_bfrag_packed(bbn + 512);
      n2 = load_bfrag_packed(bbn + 1024);
      n3 = load_bfrag_packed(bbn + 1536);
    }
    acc[0] = wmma_bf16(a, b0, acc[0]);
    acc[1] = wmma_bf16(a, b1, acc[1]);
    acc[2] = wmma_bf16(a, b2, acc[2]);
    acc[3] = wmma_bf16(a, b3, acc[3]);
    a = an; b0 = n0; b1 = n1; b2 = n2; b3 = n3;
  }

  const int lane = tid & 31;
  const int nl = lane & 15, mb = (lane >> 4) << 3;
#pragma unroll
  for (int r = 0; r < 8; ++r) {
    const int row = rowBase + mb + r;
    const float dv = dinv[row];
    const float dd = dv * dv;
#pragma unroll
    for (int nt = 0; nt < 4; ++nt) {
      const int col = colBase + (nt << 4) + nl;
      const float v = acc[nt][r];
      C[(size_t)row * 256 + col] = v;
      agg[(size_t)row * 256 + col] = v * dd + bias[col];
    }
  }
}

// ---------------- GCN aggregation / BN / pooling ----------------
__global__ void k_fill(float* __restrict__ p, float v, int n) {
  int i = blockIdx.x * blockDim.x + threadIdx.x;
  if (i < n) p[i] = v;
}
__global__ void k_deg_edges(const int* __restrict__ dst, float* __restrict__ deg) {
  int e = blockIdx.x * blockDim.x + threadIdx.x;
  if (e < kE) atomicAdd(&deg[dst[e]], 1.f);
}
__global__ void k_dinv(const float* __restrict__ deg, float* __restrict__ dinv) {
  int i = blockIdx.x * blockDim.x + threadIdx.x;
  if (i < kN) dinv[i] = __frsqrt_rn(deg[i]);
}
__global__ void k_agg_edges(const int* __restrict__ src, const int* __restrict__ dst,
                            const float* __restrict__ dinv, const float* __restrict__ h,
                            float* __restrict__ agg) {
  int idx = blockIdx.x * blockDim.x + threadIdx.x;     // E * 32 (8 channels each)
  if (idx >= kE * 32) return;
  int e = idx >> 5, q = (idx & 31) << 3;
  int s = src[e], d = dst[e];
  float co = dinv[s] * dinv[d];
  const float4* hp = (const float4*)(h + (size_t)s * 256 + q);
  const float4 h0 = hp[0], h1 = hp[1];
  float* o = agg + (size_t)d * 256 + q;
  atomicAdd(o + 0, h0.x * co);
  atomicAdd(o + 1, h0.y * co);
  atomicAdd(o + 2, h0.z * co);
  atomicAdd(o + 3, h0.w * co);
  atomicAdd(o + 4, h1.x * co);
  atomicAdd(o + 5, h1.y * co);
  atomicAdd(o + 6, h1.z * co);
  atomicAdd(o + 7, h1.w * co);
}
__global__ __launch_bounds__(256) void k_bn_stats(const float* __restrict__ x,
                                                  float* __restrict__ stats) {
  int c = threadIdx.x;
  float s = 0.f, ss = 0.f;
  for (int r = blockIdx.x; r < kN; r += gridDim.x) {
    float v = x[(size_t)r * 256 + c];
    s += v; ss += v * v;
  }
  atomicAdd(&stats[c], s);
  atomicAdd(&stats[256 + c], ss);
}
// BN + ReLU; writes fp32 (hout) + bf16 (xb). If pool != nullptr, also fuses the
// per-graph mean-pool accumulation (layer 3).
__global__ void k_bn_apply(const float* __restrict__ agg, const float* __restrict__ stats,
                           const float* __restrict__ g, const float* __restrict__ be,
                           float* __restrict__ hout, unsigned short* __restrict__ xb,
                           const int* __restrict__ bid, float* __restrict__ pool) {
  int i = blockIdx.x * blockDim.x + threadIdx.x;
  if (i >= kN * 256) return;
  int r = i >> 8, c = i & 255;
  const float invN = 1.f / (float)kN;
  float mean = stats[c] * invN;
  float var = stats[256 + c] * invN - mean * mean;
  float y = (agg[i] - mean) * __frsqrt_rn(var + kEPS) * g[c] + be[c];
  y = fmaxf(y, 0.f);
  hout[i] = y;
  xb[i] = f32_to_bf16(y);
  if (pool) atomicAdd(&pool[bid[r] * 256 + c], y);
}
__global__ void k_pool_cnt(const int* __restrict__ bid, float* __restrict__ cnt) {
  int i = blockIdx.x * blockDim.x + threadIdx.x;
  if (i < kN) atomicAdd(&cnt[bid[i]], 1.f);
}
__global__ void k_pool_div(float* __restrict__ seqs, const float* __restrict__ cnt) {
  int i = blockIdx.x * blockDim.x + threadIdx.x;
  if (i >= kB * 256) return;
  seqs[i] /= fmaxf(cnt[i >> 8], 1.f);
}

// ---------------- LSTM (WMMA gates) + classifier, single workgroup ----------------
static __device__ inline void lstm_gates(const unsigned short* xb, int Kx,
                                         const unsigned short* hb,
                                         const unsigned short* WihT,
                                         const unsigned short* WhhT,
                                         float* gates) {
  const int w = threadIdx.x >> 5;
  const int lane = threadIdx.x & 31;
  const int nl = lane & 15, mb = (lane >> 4) << 3;
  for (int t = w; t < 64; t += 8) {           // uniform: every wave runs 8 tiles
    int mi = t >> 5, ni = t & 31;
    f32x8 acc = {};
    for (int k0 = 0; k0 < Kx; k0 += 32) {
      bf16x16 a = load_afrag(xb + (mi << 4) * Kx + k0, Kx);
      bf16x16 b = load_bfrag_packed(WihT + (size_t)(k0 >> 5) * 16384 + (size_t)ni * 512);
      acc = wmma_bf16(a, b, acc);
    }
    for (int k0 = 0; k0 < 128; k0 += 32) {
      bf16x16 a = load_afrag(hb + (mi << 4) * 128 + k0, 128);
      bf16x16 b = load_bfrag_packed(WhhT + (size_t)(k0 >> 5) * 16384 + (size_t)ni * 512);
      acc = wmma_bf16(a, b, acc);
    }
#pragma unroll
    for (int r = 0; r < 8; ++r)
      gates[((mi << 4) + mb + r) * 512 + (ni << 4) + nl] = acc[r];
  }
}
static __device__ inline void lstm_cell(const float* gates, const float* bih,
                                        const float* bhh, float* cpriv,
                                        unsigned short* hb) {
  const int tid = threadIdx.x;
#pragma unroll
  for (int k = 0; k < 16; ++k) {
    int i = tid + (k << 8);                   // 4096 = 32*128 elements
    int b = i >> 7, j = i & 127;
    const float* gb = gates + (b << 9);
    float gi = gb[j]       + bih[j]       + bhh[j];
    float gf = gb[128 + j] + bih[128 + j] + bhh[128 + j];
    float gg = gb[256 + j] + bih[256 + j] + bhh[256 + j];
    float go = gb[384 + j] + bih[384 + j] + bhh[384 + j];
    float cv = sigmf(gf) * cpriv[k] + sigmf(gi) * tanhf(gg);
    float hv = sigmf(go) * tanhf(cv);
    cpriv[k] = cv;
    hb[i] = f32_to_bf16(hv);
  }
}

__global__ __launch_bounds__(256) void k_lstm(
    const float* __restrict__ seq,
    const unsigned short* __restrict__ wihT0, const unsigned short* __restrict__ whhT0,
    const unsigned short* __restrict__ wihT1, const unsigned short* __restrict__ whhT1,
    const float* __restrict__ bih0, const float* __restrict__ bhh0,
    const float* __restrict__ bih1, const float* __restrict__ bhh1,
    const float* __restrict__ Wc1, const float* __restrict__ bc1,
    const float* __restrict__ Wc2, const float* __restrict__ bc2,
    float* __restrict__ out) {
  __shared__ __align__(16) unsigned short xb[32 * 256];
  __shared__ __align__(16) unsigned short hb0[32 * 128];
  __shared__ __align__(16) unsigned short hb1[32 * 128];
  __shared__ float gates[32 * 512];
  __shared__ float feat[32 * 64];
  const int tid = threadIdx.x;
  float c0p[16], c1p[16];
#pragma unroll
  for (int k = 0; k < 16; ++k) { c0p[k] = 0.f; c1p[k] = 0.f; }
  for (int i = tid; i < 32 * 128; i += 256) { hb0[i] = 0; hb1[i] = 0; }
  __syncthreads();

  for (int s = 0; s < kS; ++s) {
    const float* x = seq + (size_t)s * kB * 256;
    for (int i = tid; i < 32 * 256; i += 256) xb[i] = f32_to_bf16(x[i]);
    __syncthreads();
    lstm_gates(xb, 256, hb0, wihT0, whhT0, gates);
    __syncthreads();
    lstm_cell(gates, bih0, bhh0, c0p, hb0);   // hb0 <- layer0 h_s
    __syncthreads();
    lstm_gates(hb0, 128, hb1, wihT1, whhT1, gates);
    __syncthreads();
    lstm_cell(gates, bih1, bhh1, c1p, hb1);   // hb1 <- layer1 h_s
    __syncthreads();
  }
  // classifier: feat = relu(hT @ Wc1^T + bc1); out = feat @ Wc2^T + bc2
  for (int i = tid; i < 32 * 64; i += 256) {
    int b = i >> 6, j = i & 63;
    float a = bc1[j];
    for (int k = 0; k < 128; ++k)
      a += bf16_to_f32(hb1[(b << 7) + k]) * Wc1[(j << 7) + k];
    feat[i] = fmaxf(a, 0.f);
  }
  __syncthreads();
  for (int i = tid; i < 96; i += 256) {
    int b = i / 3, j = i - b * 3;
    float a = bc2[j];
    for (int k = 0; k < 64; ++k) a += feat[(b << 6) + k] * Wc2[j * 64 + k];
    out[i] = a;
  }
}

// ---------------- host orchestration ----------------
extern "C" void kernel_launch(void* const* d_in, const int* in_sizes, int n_in,
                              void* d_out, int out_size, void* d_ws, size_t ws_size,
                              hipStream_t stream) {
  const float* poss = (const float*)d_in[0];
  const int*   xidx = (const int*)d_in[1];
  const int*   eidx = (const int*)d_in[2];
  const int*   bids = (const int*)d_in[3];
  const float* emb0 = (const float*)d_in[4];
  const float* emb1 = (const float*)d_in[5];
  const float* emb2 = (const float*)d_in[6];
  const float* emb5 = (const float*)d_in[7];
  const float* W1 = (const float*)d_in[8];  const float* b1 = (const float*)d_in[9];
  const float* W2 = (const float*)d_in[10]; const float* b2 = (const float*)d_in[11];
  const float* W3 = (const float*)d_in[12]; const float* b3 = (const float*)d_in[13];
  const float* g1 = (const float*)d_in[14]; const float* be1 = (const float*)d_in[15];
  const float* g2 = (const float*)d_in[16]; const float* be2 = (const float*)d_in[17];
  const float* g3 = (const float*)d_in[18]; const float* be3 = (const float*)d_in[19];
  const float* Wih0 = (const float*)d_in[20]; const float* Whh0 = (const float*)d_in[21];
  const float* bih0 = (const float*)d_in[22]; const float* bhh0 = (const float*)d_in[23];
  const float* Wih1 = (const float*)d_in[24]; const float* Whh1 = (const float*)d_in[25];
  const float* bih1 = (const float*)d_in[26]; const float* bhh1 = (const float*)d_in[27];
  const float* Wc1 = (const float*)d_in[28]; const float* bc1 = (const float*)d_in[29];
  const float* Wc2 = (const float*)d_in[30]; const float* bc2 = (const float*)d_in[31];

  char* base = (char*)d_ws;
  size_t off = 0;
  auto alloc = [&](size_t bytes) -> char* {
    char* p = base + off;
    off += (bytes + 255) & ~(size_t)255;
    return p;
  };
  unsigned short* wb1   = (unsigned short*)alloc((size_t)kK1 * 256 * 2);
  unsigned short* wb2   = (unsigned short*)alloc((size_t)256 * 256 * 2);
  unsigned short* wb3   = (unsigned short*)alloc((size_t)256 * 256 * 2);
  unsigned short* wihT0 = (unsigned short*)alloc((size_t)256 * 512 * 2);
  unsigned short* whhT0 = (unsigned short*)alloc((size_t)128 * 512 * 2);
  unsigned short* wihT1 = (unsigned short*)alloc((size_t)128 * 512 * 2);
  unsigned short* whhT1 = (unsigned short*)alloc((size_t)128 * 512 * 2);
  unsigned short* xb160 = (unsigned short*)alloc((size_t)kN * kK1 * 2);
  unsigned short* xb256 = (unsigned short*)alloc((size_t)kN * 256 * 2);
  float* h    = (float*)alloc((size_t)kN * 256 * 4);
  float* agg  = (float*)alloc((size_t)kN * 256 * 4);
  float* deg  = (float*)alloc((size_t)kN * 4);
  float* dinv = (float*)alloc((size_t)kN * 4);
  float* stats = (float*)alloc(512 * 4);
  float* seqb = (float*)alloc((size_t)kS * kB * 256 * 4);
  float* cnt  = (float*)alloc(kB * 4);
  (void)ws_size; (void)n_in; (void)in_sizes; (void)out_size;

  const int T = 256;
  auto G = [](long long n, int t) { return (unsigned)((n + t - 1) / t); };

  // weight conversion into packed WMMA-fragment layouts (deterministic each call)
  k_convert_pad_packed<<<G((long long)kK1 * 256, T), T, 0, stream>>>(W1, wb1, 129, kK1);
  k_convert_pad_packed<<<G(256LL * 256, T), T, 0, stream>>>(W2, wb2, 256, 256);
  k_convert_pad_packed<<<G(256LL * 256, T), T, 0, stream>>>(W3, wb3, 256, 256);
  k_transpose_packed<<<G(512LL * 256, T), T, 0, stream>>>(Wih0, wihT0, 512, 256);
  k_transpose_packed<<<G(512LL * 128, T), T, 0, stream>>>(Whh0, whhT0, 512, 128);
  k_transpose_packed<<<G(512LL * 128, T), T, 0, stream>>>(Wih1, wihT1, 512, 128);
  k_transpose_packed<<<G(512LL * 128, T), T, 0, stream>>>(Whh1, whhT1, 512, 128);

  const dim3 gemmGrid(kN / 128, 4);   // 256 x 4 blocks; wave-level 16x64 tiles
  for (int s = 0; s < kS; ++s) {
    const float* poss_s = poss + (size_t)s * kN;
    const int* xidx_s = xidx + (size_t)s * kN * 4;
    const int* src = eidx + (size_t)s * 2 * kE;
    const int* dst = src + kE;
    const int* bid_s = bids + (size_t)s * kN;
    float* seq_s = seqb + (size_t)s * kB * 256;

    k_assemble<<<G((long long)kN * kK1, T), T, 0, stream>>>(poss_s, xidx_s, emb0, emb1,
                                                            emb2, emb5, xb160);
    // degree (shared by all 3 layers this step)
    k_fill<<<G(kN, T), T, 0, stream>>>(deg, 1.f, kN);          // self loop
    k_deg_edges<<<G(kE, T), T, 0, stream>>>(dst, deg);
    k_dinv<<<G(kN, T), T, 0, stream>>>(deg, dinv);
    // pooling accumulators (filled by fused bn_apply of layer 3)
    k_fill<<<G(kB * 256, T), T, 0, stream>>>(seq_s, 0.f, kB * 256);
    k_fill<<<G(kB, T), T, 0, stream>>>(cnt, 0.f, kB);
    k_pool_cnt<<<G(kN, T), T, 0, stream>>>(bid_s, cnt);

    const unsigned short* xin = xb160;
    int Kin = kK1;
    const float* bl[3]  = { b1, b2, b3 };
    const float* gl[3]  = { g1, g2, g3 };
    const float* bel[3] = { be1, be2, be3 };
    const unsigned short* wl[3] = { wb1, wb2, wb3 };
    for (int l = 0; l < 3; ++l) {
      // GEMM with fused agg seed (self-loop + bias)
      k_gemm_bf16<<<gemmGrid, T, 0, stream>>>(xin, wl[l], h, agg, dinv, bl[l], Kin);
      k_agg_edges<<<G((long long)kE * 32, T), T, 0, stream>>>(src, dst, dinv, h, agg);
      k_fill<<<G(512, T), T, 0, stream>>>(stats, 0.f, 512);
      k_bn_stats<<<256, T, 0, stream>>>(agg, stats);
      k_bn_apply<<<G((long long)kN * 256, T), T, 0, stream>>>(
          agg, stats, gl[l], bel[l], h, xb256, bid_s,
          (l == 2) ? seq_s : (float*)nullptr);
      xin = xb256;
      Kin = 256;
    }
    k_pool_div<<<G(kB * 256, T), T, 0, stream>>>(seq_s, cnt);
  }

  k_lstm<<<1, T, 0, stream>>>(seqb, wihT0, whhT0, wihT1, whhT1, bih0, bhh0, bih1, bhh1,
                              Wc1, bc1, Wc2, bc2, (float*)d_out);
}
